// HAN_42408507080883
// MI455X (gfx1250) — matrix operand order repaired and checked by reading, not compiled, not run on hardware
//
#include <hip/hip_runtime.h>
#include <math.h>

typedef __attribute__((ext_vector_type(16))) _Float16 v16h;
typedef __attribute__((ext_vector_type(8)))  float    v8f;
typedef __attribute__((ext_vector_type(4)))  float    v4f;

union AFrag { v16h v; _Float16 h[16]; };
union CFrag { v8f  v; float     f[8];  };

// ---- ordered-uint encoding so float min/max can use integer atomics ----
__device__ __forceinline__ unsigned f2ord(float f) {
    unsigned u = __float_as_uint(f);
    return (u & 0x80000000u) ? ~u : (u | 0x80000000u);
}
__device__ __forceinline__ float ord2f(unsigned u) {
    return (u & 0x80000000u) ? __uint_as_float(u & 0x7FFFFFFFu)
                             : __uint_as_float(~u);
}
__device__ __forceinline__ float lrelu02(float x) { return x > 0.f ? x : 0.2f * x; }

// ---------------- init: zero accumulators, seed min/max ----------------
__global__ void k_zero(float* __restrict__ base, size_t n, unsigned* __restrict__ mm) {
    size_t i = (size_t)blockIdx.x * blockDim.x + threadIdx.x;
    size_t stride = (size_t)gridDim.x * blockDim.x;
    for (; i < n; i += stride) base[i] = 0.f;
    if (blockIdx.x == 0 && threadIdx.x == 0) {
        mm[0] = 0u;          // max accumulator (ordered-uint: smallest)
        mm[1] = 0xFFFFFFFFu; // min accumulator (ordered-uint: largest)
    }
}

// ---------------- global min/max of x_protein ----------------
__global__ void k_minmax(const float* __restrict__ x, size_t n, unsigned* __restrict__ mm) {
    __shared__ unsigned smx[256], smn[256];
    unsigned lmx = 0u, lmn = 0xFFFFFFFFu;
    size_t i = (size_t)blockIdx.x * blockDim.x + threadIdx.x;
    size_t stride = (size_t)gridDim.x * blockDim.x;
    for (; i < n; i += stride) {
        unsigned u = f2ord(x[i]);
        lmx = u > lmx ? u : lmx;
        lmn = u < lmn ? u : lmn;
    }
    int t = threadIdx.x;
    smx[t] = lmx; smn[t] = lmn;
    __syncthreads();
    for (int s = 128; s > 0; s >>= 1) {
        if (t < s) {
            smx[t] = smx[t] > smx[t + s] ? smx[t] : smx[t + s];
            smn[t] = smn[t] < smn[t + s] ? smn[t] : smn[t + s];
        }
        __syncthreads();
    }
    if (t == 0) { atomicMax(&mm[0], smx[0]); atomicMin(&mm[1], smn[0]); }
}

// ---------------- prep: transpose+convert three 64x64 weights to f16 [n][k] ----------------
__global__ void k_prep_w(const float* __restrict__ W0, const float* __restrict__ W1,
                         const float* __restrict__ W2, _Float16* __restrict__ dst) {
    int t = blockIdx.x * blockDim.x + threadIdx.x;
    if (t >= 3 * 4096) return;
    int wsel = t >> 12;
    int r = t & 4095;
    int k = r >> 6, n = r & 63;
    const float* src = (wsel == 0) ? W0 : ((wsel == 1) ? W1 : W2);
    dst[wsel * 4096 + n * 64 + k] = (_Float16)src[k * 64 + n];
}

// ---------------- WMMA projection: out[M,64] = act(A)[M,64] @ W + b ----------------
// One wave computes a full 16x64 strip: 4 N-tiles x (K=64 as two 16x16x32 WMMAs).
// A loads: four global_load_b128 per k-step; B loads: one contiguous v16h per tile.
__global__ void k_proj(const float* __restrict__ A, const _Float16* __restrict__ Wth,
                       const float* __restrict__ bias, float* __restrict__ out,
                       int M, const unsigned* __restrict__ mm, int normalize) {
    int wave = (int)(((size_t)blockIdx.x * blockDim.x + threadIdx.x) >> 5);
    int lane = threadIdx.x & 31;
    int mtiles = (M + 15) >> 4;
    if (wave >= mtiles) return;          // wave-uniform guard

    float scale = 1.f, shift = 0.f;
    if (normalize) {
        float mx = ord2f(mm[0]), mn = ord2f(mm[1]);
        float inv = 1.f / (mx - mn);
        scale = inv; shift = -mn * inv;
    }
    int hi = lane >> 4;                  // 0/1 lane group
    int ml = lane & 15;
    int m  = wave * 16 + ml;
    int mc = (m < M) ? m : (M - 1);      // clamp: no divergent loads

    CFrag c[4];
    #pragma unroll
    for (int nt = 0; nt < 4; ++nt)
        #pragma unroll
        for (int r = 0; r < 8; ++r) c[nt].f[r] = 0.f;

    const float* arow = A + (size_t)mc * 64;
    #pragma unroll
    for (int ks = 0; ks < 2; ++ks) {
        // A 16x32 f16 fragment: halves j=0..15 -> K=base..base+15 (contiguous)
        int base = hi * 16 + ks * 32;
        v4f a0 = *(const v4f*)(arow + base);
        v4f a1 = *(const v4f*)(arow + base + 4);
        v4f a2 = *(const v4f*)(arow + base + 8);
        v4f a3 = *(const v4f*)(arow + base + 12);
        AFrag a;
        #pragma unroll
        for (int j = 0; j < 4; ++j) {
            a.h[j]      = (_Float16)fmaf(a0[j], scale, shift);
            a.h[j + 4]  = (_Float16)fmaf(a1[j], scale, shift);
            a.h[j + 8]  = (_Float16)fmaf(a2[j], scale, shift);
            a.h[j + 12] = (_Float16)fmaf(a3[j], scale, shift);
        }
        #pragma unroll
        for (int nt = 0; nt < 4; ++nt) {
            int n = nt * 16 + ml;
            AFrag b;                     // B 32x16: K = j + 16*hi (+ks*32), contiguous in Wth[n][k]
            b.v = *(const v16h*)(Wth + (size_t)n * 64 + hi * 16 + ks * 32);
            c[nt].v = __builtin_amdgcn_wmma_f32_16x16x32_f16(false, a.v, false, b.v,
                                                             (short)0, c[nt].v, false, false);
        }
    }
    #pragma unroll
    for (int nt = 0; nt < 4; ++nt) {
        int n = nt * 16 + ml;
        float bv = bias[n];
        #pragma unroll
        for (int r = 0; r < 8; ++r) {
            int row = wave * 16 + r + hi * 8;    // C/D layout: M = r + 8*hi
            if (row < M) out[(size_t)row * 64 + n] = c[nt].f[r] + bv;
        }
    }
}

// ---------------- WMMA semantic GEMM: colsum[n] += sum_rows tanh((O@Wk+bk)[row,n]) ----------------
__global__ void k_tanh_colsum(const float* __restrict__ O, const _Float16* __restrict__ Wth,
                              const float* __restrict__ bk, float* __restrict__ colsum, int M) {
    int wave = (int)(((size_t)blockIdx.x * blockDim.x + threadIdx.x) >> 5);
    int lane = threadIdx.x & 31;
    int mtiles = (M + 15) >> 4;
    if (wave >= mtiles) return;
    int hi = lane >> 4;
    int ml = lane & 15;
    int m  = wave * 16 + ml;
    int mc = (m < M) ? m : (M - 1);

    CFrag c[4];
    #pragma unroll
    for (int nt = 0; nt < 4; ++nt)
        #pragma unroll
        for (int r = 0; r < 8; ++r) c[nt].f[r] = 0.f;

    const float* arow = O + (size_t)mc * 64;
    #pragma unroll
    for (int ks = 0; ks < 2; ++ks) {
        int base = hi * 16 + ks * 32;
        v4f a0 = *(const v4f*)(arow + base);
        v4f a1 = *(const v4f*)(arow + base + 4);
        v4f a2 = *(const v4f*)(arow + base + 8);
        v4f a3 = *(const v4f*)(arow + base + 12);
        AFrag a;
        #pragma unroll
        for (int j = 0; j < 4; ++j) {
            a.h[j]      = (_Float16)a0[j];
            a.h[j + 4]  = (_Float16)a1[j];
            a.h[j + 8]  = (_Float16)a2[j];
            a.h[j + 12] = (_Float16)a3[j];
        }
        #pragma unroll
        for (int nt = 0; nt < 4; ++nt) {
            int n = nt * 16 + ml;
            AFrag b;
            b.v = *(const v16h*)(Wth + (size_t)n * 64 + hi * 16 + ks * 32);
            c[nt].v = __builtin_amdgcn_wmma_f32_16x16x32_f16(false, a.v, false, b.v,
                                                             (short)0, c[nt].v, false, false);
        }
    }
    #pragma unroll
    for (int nt = 0; nt < 4; ++nt) {
        int n = nt * 16 + ml;
        float bv = bk[n];
        float acc = 0.f;
        #pragma unroll
        for (int r = 0; r < 8; ++r) {
            int row = wave * 16 + r + hi * 8;
            float t = tanhf(c[nt].f[r] + bv);
            acc += (row < M) ? t : 0.f;
        }
        atomicAdd(&colsum[n], acc);      // all 8 values in a lane share column n
    }
}

// ---------------- per-node attention logits: alpha[n,h] = sum_d x[n,h,d]*a[h,d] ----------------
__global__ void k_alpha(const float* __restrict__ x, const float* __restrict__ a,
                        float* __restrict__ alpha, int N) {
    int t = blockIdx.x * blockDim.x + threadIdx.x;
    if (t >= N * 4) return;
    int n = t >> 2, h = t & 3;
    const float* xp = x + (size_t)n * 64 + h * 16;
    const float* ap = a + h * 16;
    float s = 0.f;
    #pragma unroll
    for (int d4 = 0; d4 < 4; ++d4) {
        v4f xv = *(const v4f*)(xp + d4 * 4);
        v4f av = *(const v4f*)(ap + d4 * 4);
        s += xv[0] * av[0] + xv[1] * av[1] + xv[2] * av[2] + xv[3] * av[3];
    }
    alpha[t] = s;
}

// ---------------- edge pass 1: segment max (ordered-uint atomicMax) ----------------
__global__ void k_edge_max(const int* __restrict__ ei, int E,
                           const float* __restrict__ asrc, const float* __restrict__ adst,
                           unsigned* __restrict__ m) {
    int e = blockIdx.x * blockDim.x + threadIdx.x;
    if (e >= E) return;
    int row = ei[e], col = ei[E + e];
    v4f as = *(const v4f*)(asrc + (size_t)row * 4);
    v4f ad = *(const v4f*)(adst + (size_t)col * 4);
    #pragma unroll
    for (int h = 0; h < 4; ++h)
        atomicMax(&m[(size_t)col * 4 + h], f2ord(lrelu02(as[h] + ad[h])));
}

// ---------------- edge pass 2: z = segment_sum(exp(alpha - m)) ----------------
__global__ void k_edge_z(const int* __restrict__ ei, int E,
                         const float* __restrict__ asrc, const float* __restrict__ adst,
                         const unsigned* __restrict__ m, float* __restrict__ z) {
    int e = blockIdx.x * blockDim.x + threadIdx.x;
    if (e >= E) return;
    int row = ei[e], col = ei[E + e];
    v4f as = *(const v4f*)(asrc + (size_t)row * 4);
    v4f ad = *(const v4f*)(adst + (size_t)col * 4);
    #pragma unroll
    for (int h = 0; h < 4; ++h) {
        float al = lrelu02(as[h] + ad[h]);
        float mv = ord2f(m[(size_t)col * 4 + h]);
        atomicAdd(&z[(size_t)col * 4 + h], __expf(al - mv));
    }
}

// ---------------- edge pass 3: out[col,c] += src[row,c] * attn(e, c/16) ----------------
__global__ void k_edge_scatter(const int* __restrict__ ei, int E,
                               const float* __restrict__ asrc, const float* __restrict__ adst,
                               const unsigned* __restrict__ m, const float* __restrict__ z,
                               const float* __restrict__ src, float* __restrict__ out) {
    size_t t = (size_t)blockIdx.x * blockDim.x + threadIdx.x;
    if (t >= (size_t)E * 64) return;
    int e = (int)(t >> 6);
    int c = (int)(t & 63);
    int h = c >> 4;
    int row = ei[e], col = ei[E + e];
    float al = lrelu02(asrc[(size_t)row * 4 + h] + adst[(size_t)col * 4 + h]);
    float mv = ord2f(m[(size_t)col * 4 + h]);
    float attn = __expf(al - mv) / (z[(size_t)col * 4 + h] + 1e-16f);
    atomicAdd(&out[(size_t)col * 64 + c], src[(size_t)row * 64 + c] * attn);
}

// ---------------- elementwise relu ----------------
__global__ void k_relu(float* __restrict__ p, size_t n) {
    size_t i = (size_t)blockIdx.x * blockDim.x + threadIdx.x;
    size_t stride = (size_t)gridDim.x * blockDim.x;
    for (; i < n; i += stride) { float v = p[i]; p[i] = v > 0.f ? v : 0.f; }
}

// ---------------- semantic softmax over the 2 metapaths ----------------
__global__ void k_beta(const float* __restrict__ colsum, const float* __restrict__ q,
                       float* __restrict__ beta, float invN) {
    __shared__ float s0[64], s1[64];
    int c = threadIdx.x;                  // 64 threads
    s0[c] = q[c] * colsum[c]      * invN;
    s1[c] = q[c] * colsum[64 + c] * invN;
    __syncthreads();
    for (int s = 32; s > 0; s >>= 1) {
        if (c < s) { s0[c] += s0[c + s]; s1[c] += s1[c + s]; }
        __syncthreads();
    }
    if (c == 0) {
        float sc0 = s0[0], sc1 = s1[0];
        float mx = sc0 > sc1 ? sc0 : sc1;
        float e0 = __expf(sc0 - mx), e1 = __expf(sc1 - mx);
        float inv = 1.f / (e0 + e1);
        beta[0] = e0 * inv;
        beta[1] = e1 * inv;
    }
}

// ---------------- final classifier: d_out = (b0*o0 + b1*o1) @ W_out + b_out ----------------
__global__ void k_final(const float* __restrict__ o0, const float* __restrict__ o1,
                        const float* __restrict__ beta, const float* __restrict__ Wout,
                        const float* __restrict__ bout, float* __restrict__ out, int N) {
    int n = blockIdx.x * blockDim.x + threadIdx.x;
    if (n >= N) return;
    float b0 = beta[0], b1 = beta[1];
    float a0 = bout[0], a1 = bout[1];
    const float* p0 = o0 + (size_t)n * 64;
    const float* p1 = o1 + (size_t)n * 64;
    #pragma unroll
    for (int c4 = 0; c4 < 16; ++c4) {
        v4f u0 = *(const v4f*)(p0 + c4 * 4);
        v4f u1 = *(const v4f*)(p1 + c4 * 4);
        #pragma unroll
        for (int j = 0; j < 4; ++j) {
            float v = b0 * u0[j] + b1 * u1[j];
            int c = c4 * 4 + j;
            a0 += v * Wout[c * 2 + 0];
            a1 += v * Wout[c * 2 + 1];
        }
    }
    out[(size_t)n * 2 + 0] = a0;
    out[(size_t)n * 2 + 1] = a1;
}

extern "C" void kernel_launch(void* const* d_in, const int* in_sizes, int n_in,
                              void* d_out, int out_size, void* d_ws, size_t ws_size,
                              hipStream_t stream) {
    const float* x_gene    = (const float*)d_in[0];
    const float* x_protein = (const float*)d_in[1];
    const float* W_gene    = (const float*)d_in[2];
    const float* b_gene    = (const float*)d_in[3];
    const float* W_protein = (const float*)d_in[4];
    const float* b_protein = (const float*)d_in[5];
    const float* a_src_p2g = (const float*)d_in[6];
    const float* a_dst_p2g = (const float*)d_in[7];
    const float* a_src_g2g = (const float*)d_in[8];
    const float* a_dst_g2g = (const float*)d_in[9];
    const float* q         = (const float*)d_in[10];
    const float* Wk        = (const float*)d_in[11];
    const float* bk        = (const float*)d_in[12];
    const float* W_out     = (const float*)d_in[13];
    const float* b_out     = (const float*)d_in[14];
    const int*   ei_p2g    = (const int*)d_in[15];
    const int*   ei_g2g    = (const int*)d_in[16];

    const int NG = in_sizes[0] / 64;
    const int NP = in_sizes[1] / 64;
    const int E  = in_sizes[15] / 2;

    float* w = (float*)d_ws;
    // workspace layout (floats; all offsets multiples of 4 -> 16B aligned)
    size_t o_out0   = 0;
    size_t o_out1   = o_out0 + (size_t)NG * 64;
    size_t o_m0     = o_out1 + (size_t)NG * 64;
    size_t o_z0     = o_m0   + (size_t)NG * 4;
    size_t o_m1     = o_z0   + (size_t)NG * 4;
    size_t o_z1     = o_m1   + (size_t)NG * 4;
    size_t o_colsum = o_z1   + (size_t)NG * 4;
    size_t o_beta   = o_colsum + 128;        // end of zeroed region
    size_t o_mm     = o_beta + 16;
    size_t o_wth    = o_mm   + 16;           // 3 * 4096 halves = 6144 floats
    size_t o_hg     = o_wth  + 6144;
    size_t o_hp     = o_hg   + (size_t)NG * 64;
    size_t o_as0    = o_hp   + (size_t)NP * 64;   // alpha_src p2g (proteins)
    size_t o_ad0    = o_as0  + (size_t)NP * 4;    // alpha_dst p2g (genes)
    size_t o_as1    = o_ad0  + (size_t)NG * 4;    // alpha_src g2g (genes)
    size_t o_ad1    = o_as1  + (size_t)NG * 4;    // alpha_dst g2g (genes)

    unsigned*  mm  = (unsigned*)(w + o_mm);
    _Float16*  wth = (_Float16*)(w + o_wth);      // [0]=W_gene^T, [4096]=W_protein^T, [8192]=Wk^T

    // 1) zero accumulators + seed min/max; transpose/convert weights
    k_zero  <<<4096, 256, 0, stream>>>(w, o_beta, mm);
    k_prep_w<<<48,   256, 0, stream>>>(W_gene, W_protein, Wk, wth);

    // 2) min/max of x_protein
    k_minmax<<<1024, 256, 0, stream>>>(x_protein, (size_t)NP * 64, mm);

    // 3) WMMA projections: hg, hp (one wave per 16-row strip)
    {
        int waves = (NG + 15) / 16;
        int blocks = (waves * 32 + 255) / 256;
        k_proj<<<blocks, 256, 0, stream>>>(x_gene, wth, b_gene, w + o_hg, NG, mm, 0);
        waves = (NP + 15) / 16;
        blocks = (waves * 32 + 255) / 256;
        k_proj<<<blocks, 256, 0, stream>>>(x_protein, wth + 4096, b_protein, w + o_hp, NP, mm, 1);
    }

    // 4) attention logit vectors
    k_alpha<<<(NP * 4 + 255) / 256, 256, 0, stream>>>(w + o_hp, a_src_p2g, w + o_as0, NP);
    k_alpha<<<(NG * 4 + 255) / 256, 256, 0, stream>>>(w + o_hg, a_dst_p2g, w + o_ad0, NG);
    k_alpha<<<(NG * 4 + 255) / 256, 256, 0, stream>>>(w + o_hg, a_src_g2g, w + o_as1, NG);
    k_alpha<<<(NG * 4 + 255) / 256, 256, 0, stream>>>(w + o_hg, a_dst_g2g, w + o_ad1, NG);

    // 5) edge softmax + scatter
    int eb = (E + 255) / 256;
    int sb = (int)(((size_t)E * 64 + 255) / 256);
    k_edge_max    <<<eb, 256, 0, stream>>>(ei_p2g, E, w + o_as0, w + o_ad0, (unsigned*)(w + o_m0));
    k_edge_z      <<<eb, 256, 0, stream>>>(ei_p2g, E, w + o_as0, w + o_ad0, (unsigned*)(w + o_m0), w + o_z0);
    k_edge_scatter<<<sb, 256, 0, stream>>>(ei_p2g, E, w + o_as0, w + o_ad0, (unsigned*)(w + o_m0), w + o_z0,
                                           w + o_hp, w + o_out0);
    k_edge_max    <<<eb, 256, 0, stream>>>(ei_g2g, E, w + o_as1, w + o_ad1, (unsigned*)(w + o_m1));
    k_edge_z      <<<eb, 256, 0, stream>>>(ei_g2g, E, w + o_as1, w + o_ad1, (unsigned*)(w + o_m1), w + o_z1);
    k_edge_scatter<<<sb, 256, 0, stream>>>(ei_g2g, E, w + o_as1, w + o_ad1, (unsigned*)(w + o_m1), w + o_z1,
                                           w + o_hg, w + o_out1);

    // 6) relu on both metapath outputs
    k_relu<<<4096, 256, 0, stream>>>(w + o_out0, (size_t)NG * 64);
    k_relu<<<4096, 256, 0, stream>>>(w + o_out1, (size_t)NG * 64);

    // 7) semantic attention: column-sums of tanh(o @ Wk + bk) via WMMA, then beta
    {
        int waves = (NG + 15) / 16;
        int blocks = (waves * 32 + 255) / 256;
        k_tanh_colsum<<<blocks, 256, 0, stream>>>(w + o_out0, wth + 8192, bk, w + o_colsum, NG);
        k_tanh_colsum<<<blocks, 256, 0, stream>>>(w + o_out1, wth + 8192, bk, w + o_colsum + 64, NG);
    }
    k_beta<<<1, 64, 0, stream>>>(w + o_colsum, q, w + o_beta, 1.f / (float)NG);

    // 8) final classifier
    k_final<<<(NG + 255) / 256, 256, 0, stream>>>(w + o_out0, w + o_out1, w + o_beta,
                                                  W_out, b_out, (float*)d_out, NG);
}
